// CLIPWithMasks_16398185136697
// MI455X (gfx1250) — matrix operand order repaired and checked by reading, not compiled, not run on hardware
//
#include <hip/hip_runtime.h>
#include <hip/hip_bf16.h>
#include <math.h>

// ---------------------------------------------------------------------------
// CLIP ViT encoder stack for MI455X (gfx1250, wave32, WMMA).
// L=12, T=197, B=64, D=768, H=12, hd=64, FF=3072, M = T*B = 12608.
// GEMMs: v_wmma_f32_16x16x32_f16, 64x256 block tile, 8 waves, each wave owns
// a 32x64 strip (8 accumulators), double-buffered LDS, async-to-LDS staging
// (GLOBAL_LOAD_ASYNC_TO_LDS_B128 / ASYNCcnt), else pipelined ld+ds_store.
// ---------------------------------------------------------------------------

#define CT 197
#define CB 64
#define CD 768
#define CH 12
#define CHD 64
#define CFF 3072
#define CL 12
#define CM (CT * CB)   // 12608 rows, divisible by 64

typedef __attribute__((ext_vector_type(16))) _Float16 v16h;
typedef __attribute__((ext_vector_type(8)))  float    v8f;

// ---- async global->LDS (CDNA5 GLOBAL_LOAD_ASYNC_TO_LDS_B128) --------------
#if defined(__has_builtin)
#  if __has_builtin(__builtin_amdgcn_global_load_async_to_lds_b128)
#    define HAVE_ASYNC_LDS 1
#  endif
#endif
#ifndef HAVE_ASYNC_LDS
#  define HAVE_ASYNC_LDS 0
#endif

#if HAVE_ASYNC_LDS
// Builtin signature (from the compiler diagnostic): param0 is a pointer to
// 'int __attribute__((vector_size(16)))' in the device (AS1) address space,
// param1 the LDS (AS3) counterpart, then two immediate ints (offset, cpol).
typedef int async_v4i __attribute__((vector_size(16)));
typedef __attribute__((address_space(1))) async_v4i* async_gp;
typedef __attribute__((address_space(3))) async_v4i* async_lp;

__device__ __forceinline__ void cp16_async(const _Float16* g, _Float16* l) {
    // Flat shared addresses truncate to the LDS byte offset (aperture rule),
    // so build the AS(3) pointer from the low 32 bits.
    __builtin_amdgcn_global_load_async_to_lds_b128(
        (async_gp)(unsigned long long)(const void*)g,
        (async_lp)(unsigned int)(unsigned long long)(void*)l,
        0, 0);
}
__device__ __forceinline__ void wait_async_lds() {
#  if __has_builtin(__builtin_amdgcn_s_wait_asynccnt)
    __builtin_amdgcn_s_wait_asynccnt(0);
#  else
    asm volatile("s_wait_asynccnt 0x0" ::: "memory");
#  endif
}
#endif

// Load a 16-half WMMA fragment as two 16-byte chunks (both 16B aligned).
__device__ __forceinline__ v16h load2x8(const _Float16* p0, const _Float16* p1) {
    v16h r;
    ((float4*)&r)[0] = *(const float4*)p0;
    ((float4*)&r)[1] = *(const float4*)p1;
    return r;
}

// ---------------------------------------------------------------------------
// fp32 -> f16 cast (weight staging)
// ---------------------------------------------------------------------------
__launch_bounds__(256)
__global__ void cast_f16_kernel(const float* __restrict__ in,
                                _Float16* __restrict__ out, int n) {
    int i = blockIdx.x * 256 + threadIdx.x;
    if (i < n) out[i] = (_Float16)in[i];
}

// ---------------------------------------------------------------------------
// LayerNorm over D=768 (one row per block), emits f16 for the next GEMM.
// ---------------------------------------------------------------------------
__launch_bounds__(256)
__global__ void ln_f16_kernel(const float* __restrict__ x,
                              const float* __restrict__ g,
                              const float* __restrict__ bta,
                              _Float16* __restrict__ out) {
    __shared__ float s1[256];
    __shared__ float s2[256];
    const int row = blockIdx.x;
    const int tid = threadIdx.x;
    const float* xr = x + (size_t)row * CD;
    float v0 = xr[tid], v1 = xr[tid + 256], v2 = xr[tid + 512];
    s1[tid] = v0 + v1 + v2;
    s2[tid] = v0 * v0 + v1 * v1 + v2 * v2;
    __syncthreads();
    for (int o = 128; o > 0; o >>= 1) {
        if (tid < o) { s1[tid] += s1[tid + o]; s2[tid] += s2[tid + o]; }
        __syncthreads();
    }
    const float mu = s1[0] * (1.0f / CD);
    const float var = s2[0] * (1.0f / CD) - mu * mu;
    const float rs = rsqrtf(var + 1e-5f);
    _Float16* orow = out + (size_t)row * CD;
    orow[tid]       = (_Float16)((v0 - mu) * rs * g[tid]       + bta[tid]);
    orow[tid + 256] = (_Float16)((v1 - mu) * rs * g[tid + 256] + bta[tid + 256]);
    orow[tid + 512] = (_Float16)((v2 - mu) * rs * g[tid + 512] + bta[tid + 512]);
}

// ---------------------------------------------------------------------------
// WMMA GEMM: C[M,N] = A16[M,K] @ W16[N,K]^T + bias, optional QuickGELU,
// optional residual add, f16 and/or f32 output.
// Block tile 64(M) x 256(N), K-step 32, 8 waves (2x4), wave tile 32x64.
// Double-buffered LDS, one barrier per K-step.
// ---------------------------------------------------------------------------
__launch_bounds__(256)
__global__ void wmma_gemm_kernel(const _Float16* __restrict__ A,
                                 const _Float16* __restrict__ W,
                                 const float* __restrict__ bias,
                                 _Float16* __restrict__ outH,
                                 float* __restrict__ outF,
                                 const float* __restrict__ resid,
                                 int M, int N, int K, int gelu) {
    __shared__ _Float16 As[2][64 * 40];    // 64 rows x 32 K-halves, stride 40
    __shared__ _Float16 Ws[2][256 * 40];   // 256 rows x 32 K-halves

    const int tid  = threadIdx.x;
    const int lane = tid & 31;
    const int w    = tid >> 5;     // wave 0..7
    const int mw   = w >> 2;       // 0..1 : 32-row strip
    const int nw   = w & 3;        // 0..3 : 64-col strip
    const int Mb   = blockIdx.y * 64;
    const int Nb   = blockIdx.x * 256;

    const int srow = tid >> 2;            // staging row 0..63
    const int scol = (tid & 3) * 8;       // staging col 0,8,16,24

    const int cn = lane & 15;
    const int hl = lane >> 4;             // 0/1
    const int ak = hl * 8;                // A frag K sub-offset
    const int bk = hl * 8;                // B frag K sub-offset

    const int S = K / 32;
    v8f acc[2][4] = {};

    // Prologue: stage tile 0 into buffer 0.
#if HAVE_ASYNC_LDS
    cp16_async(&A[(size_t)(Mb + srow) * K + scol], &As[0][srow * 40 + scol]);
    #pragma unroll
    for (int p = 0; p < 4; ++p)
        cp16_async(&W[(size_t)(Nb + srow + p * 64) * K + scol],
                   &Ws[0][(srow + p * 64) * 40 + scol]);
    wait_async_lds();
#else
    *(float4*)&As[0][srow * 40 + scol] =
        *(const float4*)&A[(size_t)(Mb + srow) * K + scol];
    #pragma unroll
    for (int p = 0; p < 4; ++p)
        *(float4*)&Ws[0][(srow + p * 64) * 40 + scol] =
            *(const float4*)&W[(size_t)(Nb + srow + p * 64) * K + scol];
#endif
    __syncthreads();

    for (int k = 0; k < S; ++k) {
        const int cur = k & 1, nxt = cur ^ 1;
        const int kk = (k + 1) * 32;

#if HAVE_ASYNC_LDS
        if (k + 1 < S) {   // stage next tile straight into LDS (ASYNCcnt)
            cp16_async(&A[(size_t)(Mb + srow) * K + kk + scol],
                       &As[nxt][srow * 40 + scol]);
            #pragma unroll
            for (int p = 0; p < 4; ++p)
                cp16_async(&W[(size_t)(Nb + srow + p * 64) * K + kk + scol],
                           &Ws[nxt][(srow + p * 64) * 40 + scol]);
        }
#else
        float4 ra, rw0, rw1, rw2, rw3;     // issue loads early, drain after WMMA
        if (k + 1 < S) {
            ra  = *(const float4*)&A[(size_t)(Mb + srow) * K + kk + scol];
            rw0 = *(const float4*)&W[(size_t)(Nb + srow +   0) * K + kk + scol];
            rw1 = *(const float4*)&W[(size_t)(Nb + srow +  64) * K + kk + scol];
            rw2 = *(const float4*)&W[(size_t)(Nb + srow + 128) * K + kk + scol];
            rw3 = *(const float4*)&W[(size_t)(Nb + srow + 192) * K + kk + scol];
        }
#endif
        // 8 WMMAs on the current buffer.
        const _Float16* Ac = As[cur];
        const _Float16* Wc = Ws[cur];
        const int ar0 = (mw * 32 + cn) * 40;
        const int ar1 = (mw * 32 + 16 + cn) * 40;
        v16h a0 = load2x8(&Ac[ar0 + ak], &Ac[ar0 + 16 + ak]);
        v16h a1 = load2x8(&Ac[ar1 + ak], &Ac[ar1 + 16 + ak]);
        #pragma unroll
        for (int j = 0; j < 4; ++j) {
            const int nr = (nw * 64 + j * 16 + cn) * 40 + bk;
            v16h bj = load2x8(&Wc[nr], &Wc[nr + 16]);
            acc[0][j] = __builtin_amdgcn_wmma_f32_16x16x32_f16(false, a0, false, bj,
                                                              (short)0, acc[0][j], false, false);
            acc[1][j] = __builtin_amdgcn_wmma_f32_16x16x32_f16(false, a1, false, bj,
                                                              (short)0, acc[1][j], false, false);
        }

#if HAVE_ASYNC_LDS
        if (k + 1 < S) wait_async_lds();
#else
        if (k + 1 < S) {
            *(float4*)&As[nxt][srow * 40 + scol] = ra;
            *(float4*)&Ws[nxt][(srow +   0) * 40 + scol] = rw0;
            *(float4*)&Ws[nxt][(srow +  64) * 40 + scol] = rw1;
            *(float4*)&Ws[nxt][(srow + 128) * 40 + scol] = rw2;
            *(float4*)&Ws[nxt][(srow + 192) * 40 + scol] = rw3;
        }
#endif
        __syncthreads();
    }

    // Epilogue: C layout lane=(N col), VGPR r -> M row r + 8*(lane>=16)
    #pragma unroll
    for (int i = 0; i < 2; ++i) {
        #pragma unroll
        for (int j = 0; j < 4; ++j) {
            const int col = Nb + nw * 64 + j * 16 + cn;
            const float bv = bias[col];
            #pragma unroll
            for (int r = 0; r < 8; ++r) {
                const int row = Mb + mw * 32 + i * 16 + r + hl * 8;
                float v = acc[i][j][r] + bv;
                if (gelu) v = v * (1.0f / (1.0f + __expf(-1.702f * v)));
                if (resid) v += resid[(size_t)row * N + col];
                if (outF) outF[(size_t)row * N + col] = v;
                if (outH) outH[(size_t)row * N + col] = (_Float16)v;
            }
        }
    }
}

// ---------------------------------------------------------------------------
// Attention: one block per (q-tile of 64, head, batch). qkv: (M, 2304) f16 as
// [q|k|v]. Scores/softmax over 197 keys padded to 224; CLS-row mask hook;
// both QK^T and PV on WMMA. o16: (T,B,D) f16.
// ---------------------------------------------------------------------------
__launch_bounds__(256)
__global__ void attn_kernel(const _Float16* __restrict__ qkv,
                            const float* __restrict__ mask,   // (B,196)
                            _Float16* __restrict__ o16) {
    __shared__ _Float16 Vt[64 * 240];   // V transposed: [hd][key], stride 240
    __shared__ _Float16 Sb[64 * 240];   // scores -> probs (f16), stride 240

    const int tid  = threadIdx.x;
    const int lane = tid & 31;
    const int w    = tid >> 5;
    const int qt0  = blockIdx.x * 64;
    const int h    = blockIdx.y;
    const int b    = blockIdx.z;
    const int cn   = lane & 15;
    const int hl   = lane >> 4;

    // Stage V transposed (zero-pad keys 197..223)
    for (int idx = tid; idx < 224 * 64; idx += 256) {
        const int t = idx >> 6, d = idx & 63;
        _Float16 v = (_Float16)0;
        if (t < CT) v = qkv[(size_t)(t * CB + b) * (3 * CD) + 2 * CD + h * CHD + d];
        Vt[d * 240 + t] = v;
    }
    __syncthreads();

    // S = (Q @ K^T) * scale : 4 x 14 tiles of 16x16, 7 tiles per wave.
    for (int ti = w * 7; ti < w * 7 + 7; ++ti) {
        const int m = ti / 14, n = ti % 14;
        v8f acc = {};
        #pragma unroll
        for (int kk = 0; kk < CHD; kk += 32) {
            const _Float16* qp = qkv + (size_t)((qt0 + m * 16 + cn) * CB + b) * (3 * CD)
                                 + h * CHD + kk + hl * 8;
            v16h a = load2x8(qp, qp + 16);
            const _Float16* kp = qkv + (size_t)((n * 16 + cn) * CB + b) * (3 * CD)
                                 + CD + h * CHD + kk + hl * 8;
            v16h bb = load2x8(kp, kp + 16);
            acc = __builtin_amdgcn_wmma_f32_16x16x32_f16(false, a, false, bb,
                                                         (short)0, acc, false, false);
        }
        #pragma unroll
        for (int r = 0; r < 8; ++r)
            Sb[(m * 16 + r + hl * 8) * 240 + n * 16 + cn] = (_Float16)(acc[r] * 0.125f);
    }
    __syncthreads();

    // Row softmax (wave handles 8 rows), CLS mask hook on global row 0.
    for (int j = 0; j < 8; ++j) {
        const int r = w * 8 + j;
        const int qrow = qt0 + r;
        float mx = -1e30f;
        for (int c = lane; c < CT; c += 32) mx = fmaxf(mx, (float)Sb[r * 240 + c]);
        for (int o = 16; o > 0; o >>= 1) mx = fmaxf(mx, __shfl_xor(mx, o, 32));
        float sum = 0.0f;
        for (int c = lane; c < CT; c += 32) sum += __expf((float)Sb[r * 240 + c] - mx);
        for (int o = 16; o > 0; o >>= 1) sum += __shfl_xor(sum, o, 32);
        const float inv = 1.0f / sum;
        for (int c = lane; c < 224; c += 32) {
            float p = 0.0f;
            if (qrow < CT && c < CT) {
                p = __expf((float)Sb[r * 240 + c] - mx) * inv;
                if (qrow == 0 && c >= 1) p *= mask[(size_t)b * (CT - 1) + (c - 1)];
            }
            Sb[r * 240 + c] = (_Float16)p;
        }
    }
    __syncthreads();

    // O = P @ V : 4 x 4 tiles of 16x16, 2 per wave, K = 224 padded keys.
    for (int t2 = w * 2; t2 < w * 2 + 2; ++t2) {
        const int m = t2 >> 2, n = t2 & 3;
        v8f acc = {};
        #pragma unroll
        for (int kk = 0; kk < 224; kk += 32) {
            const _Float16* ap = &Sb[(m * 16 + cn) * 240 + kk + hl * 8];
            v16h a = load2x8(ap, ap + 16);
            const _Float16* vp = &Vt[(n * 16 + cn) * 240 + kk + hl * 8];
            v16h bb = load2x8(vp, vp + 16);
            acc = __builtin_amdgcn_wmma_f32_16x16x32_f16(false, a, false, bb,
                                                         (short)0, acc, false, false);
        }
        #pragma unroll
        for (int r = 0; r < 8; ++r) {
            const int qrow = qt0 + m * 16 + r + hl * 8;
            if (qrow < CT)
                o16[(size_t)(qrow * CB + b) * CD + h * CHD + n * 16 + cn] = (_Float16)acc[r];
        }
    }
}

// ---------------------------------------------------------------------------
// Host orchestration
// ---------------------------------------------------------------------------
extern "C" void kernel_launch(void* const* d_in, const int* in_sizes, int n_in,
                              void* d_out, int out_size, void* d_ws, size_t ws_size,
                              hipStream_t stream) {
    (void)in_sizes; (void)n_in; (void)out_size; (void)ws_size;

    const float* hiddens = (const float*)d_in[0];
    const float* masks   = (const float*)d_in[1];
    const float* qkv_w   = (const float*)d_in[2];
    const float* qkv_b   = (const float*)d_in[3];
    const float* out_w   = (const float*)d_in[4];
    const float* out_b   = (const float*)d_in[5];
    const float* ln1_g   = (const float*)d_in[6];
    const float* ln1_b   = (const float*)d_in[7];
    const float* ln2_g   = (const float*)d_in[8];
    const float* ln2_b   = (const float*)d_in[9];
    const float* mlp_w1  = (const float*)d_in[10];
    const float* mlp_b1  = (const float*)d_in[11];
    const float* mlp_w2  = (const float*)d_in[12];
    const float* mlp_b2  = (const float*)d_in[13];

    char* ws = (char*)d_ws;
    size_t off = 0;
    auto wsalloc = [&](size_t bytes) -> void* {
        void* p = (void*)(ws + off);
        off += (bytes + 255) & ~(size_t)255;
        return p;
    };

    float*     x     = (float*)    wsalloc((size_t)CM * CD * 4);          // residual state
    _Float16*  bufA  = (_Float16*) wsalloc((size_t)CM * CD * 2);          // h16 / o16
    _Float16*  bufB  = (_Float16*) wsalloc((size_t)CM * CFF * 2);         // qkv16 / mlp hid
    _Float16*  wq16  = (_Float16*) wsalloc((size_t)3 * CD * CD * 2);
    _Float16*  wo16  = (_Float16*) wsalloc((size_t)CD * CD * 2);
    _Float16*  w116  = (_Float16*) wsalloc((size_t)CFF * CD * 2);
    _Float16*  w216  = (_Float16*) wsalloc((size_t)CD * CFF * 2);

    (void)hipMemcpyAsync(x, hiddens, (size_t)CM * CD * 4, hipMemcpyDeviceToDevice, stream);

    const int nQKV = 3 * CD * CD;
    const int nOUT = CD * CD;
    const int nW1  = CFF * CD;
    const int nW2  = CD * CFF;

    for (int l = 0; l < CL; ++l) {
        cast_f16_kernel<<<(nQKV + 255) / 256, 256, 0, stream>>>(qkv_w + (size_t)l * nQKV, wq16, nQKV);
        cast_f16_kernel<<<(nOUT + 255) / 256, 256, 0, stream>>>(out_w + (size_t)l * nOUT, wo16, nOUT);
        cast_f16_kernel<<<(nW1  + 255) / 256, 256, 0, stream>>>(mlp_w1 + (size_t)l * nW1, w116, nW1);
        cast_f16_kernel<<<(nW2  + 255) / 256, 256, 0, stream>>>(mlp_w2 + (size_t)l * nW2, w216, nW2);

        // h = LN1(x)
        ln_f16_kernel<<<CM, 256, 0, stream>>>(x, ln1_g + l * CD, ln1_b + l * CD, bufA);
        // qkv = h @ qkv_w^T + qkv_b        (N = 2304 = 9 * 256)
        wmma_gemm_kernel<<<dim3(3 * CD / 256, CM / 64), 256, 0, stream>>>(
            bufA, wq16, qkv_b + (size_t)l * 3 * CD, bufB, nullptr, nullptr,
            CM, 3 * CD, CD, 0);
        // attention (with CLS mask hook) -> o16
        attn_kernel<<<dim3(4, CH, CB), 256, 0, stream>>>(bufB, masks, bufA);
        // x = x + o @ out_w^T + out_b      (N = 768 = 3 * 256)
        wmma_gemm_kernel<<<dim3(CD / 256, CM / 64), 256, 0, stream>>>(
            bufA, wo16, out_b + (size_t)l * CD, nullptr, x, x,
            CM, CD, CD, 0);
        // h2 = LN2(x)
        ln_f16_kernel<<<CM, 256, 0, stream>>>(x, ln2_g + l * CD, ln2_b + l * CD, bufA);
        // m = quickgelu(h2 @ w1^T + b1)    (N = 3072 = 12 * 256)
        wmma_gemm_kernel<<<dim3(CFF / 256, CM / 64), 256, 0, stream>>>(
            bufA, w116, mlp_b1 + (size_t)l * CFF, bufB, nullptr, nullptr,
            CM, CFF, CD, 1);
        // x = x + m @ w2^T + b2  (last layer writes d_out directly)
        float* xo = (l == CL - 1) ? (float*)d_out : x;
        wmma_gemm_kernel<<<dim3(CD / 256, CM / 64), 256, 0, stream>>>(
            bufB, w216, mlp_b2 + (size_t)l * CD, nullptr, xo, x,
            CM, CD, CFF, 0);
    }
}
